// Reconstructeur_18940805775490
// MI455X (gfx1250) — matrix operand order repaired and checked by reading, not compiled, not run on hardware
//
#include <hip/hip_runtime.h>

// ---------------------------------------------------------------------------
// Sizes from the reference
// ---------------------------------------------------------------------------
#define NM    16
#define G_ALL 4096
#define LAT   1024
#define S0    512
#define S1    256
#define S2    128

// Main-kernel tiling
#define GT    128              // grid rows per block
#define PY    520              // Y   LDS row pitch (halves): 260 dwords, 260%64==4
#define P1    264              // H1  LDS row pitch (halves): 132 dwords
#define P2    136              // H2  LDS row pitch (halves): 68 dwords
#define PB    40               // weight-chunk LDS row pitch (halves): 16+4 dwords (TDM pad)

#define OFF_Y   0
#define OFF_H2  0                                  // overlays Y (Y dead by stage C epilogue)
#define OFF_H1  (GT * PY * 2)                      // 133120
#define OFF_B0  (OFF_H1 + GT * P1 * 2)             // 200704: weight buffer 0
#define BUF_BYTES (S1 * PB * 2)                    // 20480 (max chunk: 256 rows x 40 halves)
#define OFF_B1  (OFF_B0 + BUF_BYTES)               // 221184: weight buffer 1
#define SMEM_BYTES (OFF_B1 + BUF_BYTES)            // 241664 (~236 KB of 320 KB/WGP)

// workspace layout (bytes)
#define WS_CVEC 0                                  // 16*512 f32 = 32 KB
#define WS_W1H  32768                              // 16*256*512 f16 = 4 MB
#define WS_W2H  (32768 + NM * S1 * S0 * 2)         // 16*128*256 f16 = 1 MB

#define AS3 __attribute__((address_space(3)))

typedef __attribute__((ext_vector_type(16))) _Float16 v16h;
typedef __attribute__((ext_vector_type(8)))  _Float16 v8h;
typedef __attribute__((ext_vector_type(4)))  _Float16 v4h;
typedef __attribute__((ext_vector_type(8)))  float    v8f;
typedef __attribute__((ext_vector_type(4)))  float    v4f;
typedef __attribute__((ext_vector_type(4)))  unsigned int u32x4;
typedef __attribute__((ext_vector_type(8)))  int      i32x8;
typedef __attribute__((ext_vector_type(4)))  int      i32x4;

// explicit-LDS types: all shared accesses go through address_space(3) pointers
typedef AS3 _Float16 lds_half;
typedef AS3 float    lds_float;
typedef AS3 v8h      lds_v8h;

union V16H { v16h v; v8h h[2]; };

static __device__ inline v8f wmma_f16(const V16H& a, const V16H& b, v8f c) {
    return __builtin_amdgcn_wmma_f32_16x16x32_f16(
        /*neg_a=*/false, a.v, /*neg_b=*/false, b.v,
        /*c_mod=*/(short)0, c, /*reuse_a=*/false, /*reuse_b=*/false);
}

// ---------------------------------------------------------------------------
// TDM: DMA a 2D f16 tile [rows x 32] (row stride `stride` elems) from global
// into LDS at byte offset lds_off, padded 4 dwords after every 16 dwords so
// the LDS row pitch is PB=40 halves. Descriptor per CDNA5 ISA ch.8 (D#).
// ---------------------------------------------------------------------------
static __device__ inline void tdm_load_w_chunk(unsigned lds_off,
                                               const _Float16* gsrc,
                                               int rows, int stride) {
    unsigned long long ga = (unsigned long long)(size_t)gsrc;
    u32x4 g0 = { 1u,                                   // count=1 (valid user D#)
                 lds_off,                              // lds_addr (bytes)
                 (unsigned)(ga & 0xFFFFFFFFu),         // global_addr[31:0]
                 (unsigned)((ga >> 32) & 0x01FFFFFFu)  // global_addr[56:32]
                   | (2u << 30) };                     // type=2 ("image")
    i32x8 g1 = { (1 << 16)                             // data_size = 2 bytes
                   | (1 << 20)                         // pad_enable
                   | (3 << 22)                         // pad_interval: 16 dwords
                   | (3 << 25),                        // pad_amount: 4 dwords
                 (stride & 0xFFFF) << 16,              // tensor_dim0[15:0]
                 ((stride >> 16) & 0xFFFF) | ((rows & 0xFFFF) << 16), // td0 hi | td1 lo
                 (32 << 16),                           // td1 hi (0) | tile_dim0 = 32
                 rows & 0xFFFF,                        // tile_dim1 | tile_dim2=0
                 stride,                               // tensor_dim0_stride[31:0]
                 0, 0 };
    i32x4 z4 = { 0, 0, 0, 0 };
    __asm__ volatile("" ::: "memory");
#if __clang_major__ >= 23
    i32x8 z8 = { 0, 0, 0, 0, 0, 0, 0, 0 };
    __builtin_amdgcn_tensor_load_to_lds(g0, g1, z4, z4, z8, 0);
#else
    __builtin_amdgcn_tensor_load_to_lds(g0, g1, z4, z4, 0);
#endif
    __asm__ volatile("" ::: "memory");
}

static __device__ inline void tdm_wait(int pending_is_one) {
    if (pending_is_one) __builtin_amdgcn_s_wait_tensorcnt(1);
    else                __builtin_amdgcn_s_wait_tensorcnt(0);
    __asm__ volatile("" ::: "memory");
}

// ---------------------------------------------------------------------------
// Kernel 0a: cvec[m][o] = Wa[m][o] . x0 + ba[m][o] + bb[m][o]     (16x512)
// ---------------------------------------------------------------------------
__global__ void __launch_bounds__(256)
precompute_c_kernel(const float* __restrict__ Wa, const float* __restrict__ x0,
                    const float* __restrict__ ba, const float* __restrict__ bb,
                    float* __restrict__ cvec) {
    const int m    = blockIdx.x;
    const int lane = threadIdx.x & 31;
    const int wave = threadIdx.x >> 5;          // 0..7

    float xv[32];
#pragma unroll
    for (int j = 0; j < 32; ++j) xv[j] = x0[j * 32 + lane];

    for (int i = 0; i < 64; ++i) {
        const int o = wave * 64 + i;
        const float* wrow = Wa + ((size_t)(m * S0 + o)) * LAT;
        float s = 0.f;
#pragma unroll
        for (int j = 0; j < 32; ++j) s = fmaf(wrow[j * 32 + lane], xv[j], s);
        for (int off = 16; off >= 1; off >>= 1) s += __shfl_xor(s, off, 32);
        if (lane == 0) cvec[m * S0 + o] = s + ba[m * S0 + o] + bb[m * S0 + o];
    }
}

// ---------------------------------------------------------------------------
// Kernel 0b: one-shot f32 -> f16 conversion of W1, W2 into workspace
// ---------------------------------------------------------------------------
__global__ void __launch_bounds__(256)
convert_weights_kernel(const float* __restrict__ W1, const float* __restrict__ W2,
                       _Float16* __restrict__ W1h, _Float16* __restrict__ W2h) {
    const int n1 = NM * S1 * S0;                  // 2,097,152 (mult of 4)
    const int n2 = NM * S2 * S1;                  //   524,288 (mult of 4)
    const int t  = (blockIdx.x * 256 + threadIdx.x) * 4;
    if (t < n1) {
        v4f f = *(const v4f*)(W1 + t);
        v4h h; h[0] = (_Float16)f[0]; h[1] = (_Float16)f[1];
               h[2] = (_Float16)f[2]; h[3] = (_Float16)f[3];
        *(v4h*)(W1h + t) = h;
    } else {
        const int u = t - n1;
        if (u < n2) {
            v4f f = *(const v4f*)(W2 + u);
            v4h h; h[0] = (_Float16)f[0]; h[1] = (_Float16)f[1];
                   h[2] = (_Float16)f[2]; h[3] = (_Float16)f[3];
            *(v4h*)(W2h + u) = h;
        }
    }
}

// ---------------------------------------------------------------------------
// Main fused decoder kernel: one block = (mlp m, 128 grid rows)
// ---------------------------------------------------------------------------
__global__ void __launch_bounds__(512)
decoder_kernel(const float* __restrict__ cvec,
               const float* __restrict__ Wb,
               const _Float16* __restrict__ W1h, const float* __restrict__ b1,
               const _Float16* __restrict__ W2h, const float* __restrict__ b2,
               const float* __restrict__ W3,     const float* __restrict__ b3,
               float* __restrict__ out) {
    // Dynamic LDS base offset (static LDS size == dynamic-LDS start offset)
    const unsigned sbase = __builtin_amdgcn_groupstaticsize();
    lds_half*  Ysh  = (lds_half*)(sbase + OFF_Y);
    lds_half*  H1sh = (lds_half*)(sbase + OFF_H1);
    lds_half*  H2sh = (lds_half*)(sbase + OFF_H2);
    lds_half*  Bsh0 = (lds_half*)(sbase + OFF_B0);
    lds_half*  Bsh1 = (lds_half*)(sbase + OFF_B1);
    lds_float* W3sh = (lds_float*)(sbase + OFF_B0);
    const unsigned boff0 = sbase + OFF_B0;
    const unsigned boff1 = sbase + OFF_B1;

    const int m    = blockIdx.x >> 5;     // 0..15
    const int gb   = blockIdx.x & 31;     // 0..31
    const int g0   = gb * GT;
    const int tid  = threadIdx.x;
    const int lane = tid & 31;
    const int wave = tid >> 5;            // 0..15
    const int gt   = wave >> 1;           // g-tile 0..7
    const int hs   = wave & 1;            // h half-range
    const int nn   = lane & 15;
    const int hi   = lane >> 4;

    const _Float16* W1m = W1h + (size_t)m * S1 * S0;
    const _Float16* W2m = W2h + (size_t)m * S2 * S1;

    // Deterministic init of weight buffers (visible LDS stores; also makes
    // behavior defined even before the first DMA lands). 512 thr x 80 B.
    {
        lds_v8h* p = (lds_v8h*)(Bsh0 + tid * 40);
        v8h z; 
#pragma unroll
        for (int i = 0; i < 8; ++i) z[i] = (_Float16)0.f;
#pragma unroll
        for (int i = 0; i < 5; ++i) p[i] = z;
    }
    __syncthreads();

    // Prefetch W1 chunk 0 via TDM: overlaps the whole of Stage A
    if (wave == 0) tdm_load_w_chunk(boff0, W1m, S1, S0);

    // ---------------- Stage A: Y = relu(c + Wb.(gx,gy)) -> LDS f16 ----------
    {
        const int o  = tid;               // 0..511
        const float c0 = cvec[m * S0 + o];
        const float w0 = Wb[(m * S0 + o) * 2 + 0];
        const float w1 = Wb[(m * S0 + o) * 2 + 1];
        const float step = 2.0f / 63.0f;
        for (int r = 0; r < GT; ++r) {
            const int g  = g0 + r;
            const float gx = fmaf(step, (float)(g & 63), -1.0f);
            const float gy = fmaf(step, (float)(g >> 6), -1.0f);
            float v = fmaf(w1, gy, fmaf(w0, gx, c0));
            v = v > 0.f ? v : 0.f;
            Ysh[r * PY + o] = (_Float16)v;
        }
    }
    __syncthreads();

    // ---------------- Stage B: H1 = relu(Y x W1^T + b1)  [128x256] ----------
    v8f acc[8];
#pragma unroll
    for (int j = 0; j < 8; ++j)
#pragma unroll
        for (int r = 0; r < 8; ++r) acc[j][r] = 0.f;

    for (int kb = 0; kb < S0 / 32; ++kb) {
        if (wave == 0) {
            if (kb < S0 / 32 - 1) {   // issue next chunk, wait for current
                tdm_load_w_chunk((kb & 1) ? boff0 : boff1, W1m + (kb + 1) * 32, S1, S0);
                tdm_wait(1);
            } else {
                tdm_wait(0);
            }
        }
        __syncthreads();

        const lds_half* Bc = (kb & 1) ? Bsh1 : Bsh0;
        V16H a;   // A fragment (16x32 f16, ISA interleaved layout)
        const lds_half* ap = Ysh + (size_t)(gt * 16 + nn) * PY + kb * 32 + hi * 8;
        a.h[0] = *(const lds_v8h*)(ap);
        a.h[1] = *(const lds_v8h*)(ap + 16);

#pragma unroll
        for (int j = 0; j < 8; ++j) {
            const int ht = hs * 8 + j;
            V16H b;   // B fragment (32x16): lane holds interleaved K (same as A) for column nn
            const lds_half* bp = Bc + (ht * 16 + nn) * PB + hi * 8;
            b.h[0] = *(const lds_v8h*)(bp);
            b.h[1] = *(const lds_v8h*)(bp + 16);
            acc[j] = wmma_f16(a, b, acc[j]);
        }
        __syncthreads();
    }

    // Prefetch W2 chunk 0 (overlaps Stage B epilogue)
    if (wave == 0) tdm_load_w_chunk(boff0, W2m, S2, S1);

    // epilogue: bias + relu -> H1 (C/D layout: vgpr r -> M=hi*8+r, N=nn)
    {
        const int mb = hi * 8;
#pragma unroll
        for (int j = 0; j < 8; ++j) {
            const int col = (hs * 8 + j) * 16 + nn;
            const float bv = b1[m * S1 + col];
#pragma unroll
            for (int r = 0; r < 8; ++r) {
                float v = acc[j][r] + bv;
                v = v > 0.f ? v : 0.f;
                H1sh[(gt * 16 + mb + r) * P1 + col] = (_Float16)v;
            }
        }
    }

    // ---------------- Stage C: H2 = relu(H1 x W2^T + b2)  [128x128] ---------
    v8f acc2[4];
#pragma unroll
    for (int j = 0; j < 4; ++j)
#pragma unroll
        for (int r = 0; r < 8; ++r) acc2[j][r] = 0.f;

    for (int kb = 0; kb < S1 / 32; ++kb) {
        if (wave == 0) {
            if (kb < S1 / 32 - 1) {
                tdm_load_w_chunk((kb & 1) ? boff0 : boff1, W2m + (kb + 1) * 32, S2, S1);
                tdm_wait(1);
            } else {
                tdm_wait(0);
            }
        }
        __syncthreads();

        const lds_half* Bc = (kb & 1) ? Bsh1 : Bsh0;
        V16H a;
        const lds_half* ap = H1sh + (size_t)(gt * 16 + nn) * P1 + kb * 32 + hi * 8;
        a.h[0] = *(const lds_v8h*)(ap);
        a.h[1] = *(const lds_v8h*)(ap + 16);

#pragma unroll
        for (int j = 0; j < 4; ++j) {
            const int ht = hs * 4 + j;
            V16H b;
            const lds_half* bp = Bc + (ht * 16 + nn) * PB + hi * 8;
            b.h[0] = *(const lds_v8h*)(bp);
            b.h[1] = *(const lds_v8h*)(bp + 16);
            acc2[j] = wmma_f16(a, b, acc2[j]);
        }
        __syncthreads();
    }
    // epilogue -> H2 (overlays dead Y region)
    {
        const int mb = hi * 8;
#pragma unroll
        for (int j = 0; j < 4; ++j) {
            const int col = (hs * 4 + j) * 16 + nn;
            const float bv = b2[m * S2 + col];
#pragma unroll
            for (int r = 0; r < 8; ++r) {
                float v = acc2[j][r] + bv;
                v = v > 0.f ? v : 0.f;
                H2sh[(gt * 16 + mb + r) * P2 + col] = (_Float16)v;
            }
        }
    }

    // stage W3 (3x128 f32) into LDS (buffers idle: tensorcnt==0 + loop barrier)
    if (tid < 3 * S2) W3sh[tid] = W3[m * 3 * S2 + tid];
    __syncthreads();

    // ---------------- Stage D: out = tanh(W3 . h2 + b3) ---------------------
    {
        const int g = tid >> 2;       // 0..127
        const int c = tid & 3;
        if (c < 3) {
            const lds_float* wr = W3sh + c * S2;
            const lds_half*  hr = H2sh + g * P2;
            float s = b3[m * 3 + c];
#pragma unroll 8
            for (int k = 0; k < S2; ++k) s = fmaf((float)hr[k], wr[k], s);
            out[((size_t)(m * G_ALL + g0 + g)) * 3 + c] = tanhf(s);
        }
    }
}

// ---------------------------------------------------------------------------
// Host launcher
// ---------------------------------------------------------------------------
extern "C" void kernel_launch(void* const* d_in, const int* in_sizes, int n_in,
                              void* d_out, int out_size, void* d_ws, size_t ws_size,
                              hipStream_t stream) {
    const float* x0 = (const float*)d_in[0];
    const float* Wa = (const float*)d_in[1];
    const float* ba = (const float*)d_in[2];
    const float* Wb = (const float*)d_in[3];
    const float* bb = (const float*)d_in[4];
    const float* W1 = (const float*)d_in[5];
    const float* b1 = (const float*)d_in[6];
    const float* W2 = (const float*)d_in[7];
    const float* b2 = (const float*)d_in[8];
    const float* W3 = (const float*)d_in[9];
    const float* b3 = (const float*)d_in[10];
    float* out = (float*)d_out;

    float*     cvec = (float*)((char*)d_ws + WS_CVEC);
    _Float16*  W1h  = (_Float16*)((char*)d_ws + WS_W1H);
    _Float16*  W2h  = (_Float16*)((char*)d_ws + WS_W2H);

    (void)in_sizes; (void)n_in; (void)out_size; (void)ws_size;

    hipFuncSetAttribute((const void*)decoder_kernel,
                        hipFuncAttributeMaxDynamicSharedMemorySize, SMEM_BYTES);

    precompute_c_kernel<<<NM, 256, 0, stream>>>(Wa, x0, ba, bb, cvec);
    const int ncvt = (NM * S1 * S0 + NM * S2 * S1) / 4;   // elems/4 per thread
    convert_weights_kernel<<<ncvt / 256, 256, 0, stream>>>(W1, W2, W1h, W2h);
    decoder_kernel<<<NM * (G_ALL / GT), 512, SMEM_BYTES, stream>>>(
        cvec, Wb, W1h, b1, W2h, b2, W3, b3, out);
}
